// GroupedQueryAttention_51110110823116
// MI455X (gfx1250) — matrix operand order repaired and checked by reading, not compiled, not run on hardware
//
#include <hip/hip_runtime.h>
#include <math.h>

// ---- problem constants (match reference) ----
#define BB     4
#define LL     1024
#define DIN    2048
#define DOUT   2048
#define HH     32
#define GG     8
#define HD     64
#define KVW    512          // G*HD

typedef __attribute__((ext_vector_type(16))) __bf16          v16bf;
typedef __attribute__((ext_vector_type(8)))  float           v8f;
typedef __attribute__((ext_vector_type(8)))  unsigned short  u16x8;

// WMMA bf16 fragment: 16 bf16 per lane.
// A-operand (16x32): lane l holds row M=l%16; u[j] = K kk+16*(l/16)+j.
// B-operand (32x16): lane l holds col N=l%16; u[j] = K kk+16*(l/16)+j.
union Frag {
  v16bf  v;
  __bf16 h[16];
  u16x8  q[2];
};

#define WMMA_BF16(a, b, c) \
  __builtin_amdgcn_wmma_f32_16x16x32_bf16(false, (a), false, (b), (short)0, (c), false, false)

// ------------------------------------------------------------------
// Flat fp32 -> bf16 conversion (8 elements / thread).
// ------------------------------------------------------------------
__global__ void gqa_cvt_bf16(const float* __restrict__ in, __bf16* __restrict__ out) {
  const size_t i = ((size_t)blockIdx.x * 256 + threadIdx.x) * 8;
  #pragma unroll
  for (int j = 0; j < 8; ++j) out[i + j] = (__bf16)in[i + j];
}

// ------------------------------------------------------------------
// Wt[n*K + k] = (bf16) W[k*N + n]   (32x32 LDS tile transpose)
// ------------------------------------------------------------------
__global__ void gqa_transpose_cvt(const float* __restrict__ W, __bf16* __restrict__ Wt,
                                  int K, int N) {
  __shared__ __bf16 tile[32][33];
  const int k0 = blockIdx.y * 32, n0 = blockIdx.x * 32;
  const int tx = threadIdx.x & 31, ty = threadIdx.x >> 5;   // 32 x 8
  #pragma unroll
  for (int i = ty; i < 32; i += 8)
    tile[i][tx] = (__bf16)W[(size_t)(k0 + i) * N + n0 + tx];
  __syncthreads();
  #pragma unroll
  for (int i = ty; i < 32; i += 8)
    Wt[(size_t)(n0 + i) * K + k0 + tx] = tile[tx][i];
}

// ------------------------------------------------------------------
// C[M,N] = A[M,K] * Bt[N,K]^T, all-bf16 operands, f32 accum.
// Per-wave macro-tile: 32(M) x 64(N) -> 8 WMMAs per K-step.
// 4 waves/block along N => block covers 32 x 256.
// store mode: 0 = fp32 row-major (Cf), 1 = bf16 row-major (Cb),
//             2 = bf16 per-batch transposed vt[(b*N+n)*LL + s] (Cb)
// ------------------------------------------------------------------
__global__ void gqa_gemm(const __bf16* __restrict__ A, const __bf16* __restrict__ Bt,
                         float* __restrict__ Cf, __bf16* __restrict__ Cb,
                         int M, int N, int K, int mode) {
  const int lane  = threadIdx.x & 31;
  const int wave  = threadIdx.x >> 5;                 // 0..3
  const int r16   = lane & 15;
  const int khalf = lane >> 4;
  const int nBase = (blockIdx.x * 4 + wave) * 64;
  const int mBase = blockIdx.y * 32;

  v8f acc[2][4] = {};
  for (int kk = 0; kk < K; kk += 32) {
    Frag a[2], b[4];
    #pragma unroll
    for (int i = 0; i < 2; ++i) {
      const __bf16* ap = A + (size_t)(mBase + i * 16 + r16) * K + kk + khalf * 16;
      a[i].q[0] = *(const u16x8*)ap;
      a[i].q[1] = *(const u16x8*)(ap + 8);
    }
    const int k0 = kk + khalf * 16;
    #pragma unroll
    for (int j = 0; j < 4; ++j) {
      const __bf16* bp = Bt + (size_t)(nBase + j * 16 + r16) * K + k0;
      b[j].q[0] = *(const u16x8*)bp;
      b[j].q[1] = *(const u16x8*)(bp + 8);
    }
    #pragma unroll
    for (int i = 0; i < 2; ++i)
      #pragma unroll
      for (int j = 0; j < 4; ++j)
        acc[i][j] = WMMA_BF16(a[i].v, b[j].v, acc[i][j]);
  }
  // epilogue: C/D layout: VGPR ii -> row = 8*khalf + ii, col = r16
  #pragma unroll
  for (int i = 0; i < 2; ++i) {
    #pragma unroll
    for (int j = 0; j < 4; ++j) {
      const int col = nBase + j * 16 + r16;
      #pragma unroll
      for (int ii = 0; ii < 8; ++ii) {
        const int row = mBase + i * 16 + 8 * khalf + ii;
        const float val = acc[i][j][ii];
        if (mode == 0) {
          Cf[(size_t)row * N + col] = val;
        } else if (mode == 1) {
          Cb[(size_t)row * N + col] = (__bf16)val;
        } else {        // vt: per-batch transpose
          Cb[((size_t)(row >> 10) * N + col) * LL + (row & (LL - 1))] = (__bf16)val;
        }
      }
    }
  }
}

// ------------------------------------------------------------------
// scores[bh,t,s] = sum_d q[b,t,h*64+d] * k[b,s,g*64+d]  (+ causal -inf)
// Per-wave: 16(t) x 64(s); contraction HD=64 (2 K-steps, 8 WMMAs).
// ------------------------------------------------------------------
__global__ void gqa_scores(const __bf16* __restrict__ q, const __bf16* __restrict__ k,
                           float* __restrict__ wout) {
  const int lane  = threadIdx.x & 31;
  const int wave  = threadIdx.x >> 5;                 // 0..7
  const int sBase = (blockIdx.x * 8 + wave) * 64;
  const int tTile = blockIdx.y;                       // 0..63
  const int bh    = blockIdx.z;                       // 0..127
  const int b     = bh >> 5;
  const int h     = bh & 31;
  const int qoff  = h * HD;
  const int koff  = (h >> 2) * HD;                    // g*HD
  const int r16   = lane & 15;
  const int khalf = lane >> 4;
  const int t     = tTile * 16 + r16;

  v8f acc[4] = {};
  if (sBase <= tTile * 16 + 15) {                     // wave-uniform causal skip
    #pragma unroll
    for (int kk = 0; kk < HD; kk += 32) {
      Frag a, bfr[4];
      const __bf16* ap = q + ((size_t)(b * LL + t)) * DOUT + qoff + kk + khalf * 16;
      a.q[0] = *(const u16x8*)ap;
      a.q[1] = *(const u16x8*)(ap + 8);
      const int k0 = kk + khalf * 16;
      #pragma unroll
      for (int j = 0; j < 4; ++j) {
        const __bf16* bp = k + ((size_t)(b * LL + sBase + j * 16 + r16)) * KVW + koff + k0;
        bfr[j].q[0] = *(const u16x8*)bp;
        bfr[j].q[1] = *(const u16x8*)(bp + 8);
      }
      #pragma unroll
      for (int j = 0; j < 4; ++j)
        acc[j] = WMMA_BF16(a.v, bfr[j].v, acc[j]);
    }
  }
  #pragma unroll
  for (int j = 0; j < 4; ++j) {
    const int sc = sBase + j * 16 + r16;
    #pragma unroll
    for (int i = 0; i < 8; ++i) {
      const int tr = tTile * 16 + 8 * khalf + i;
      wout[((size_t)bh * LL + tr) * LL + sc] = (sc > tr) ? -INFINITY : acc[j][i];
    }
  }
}

// ------------------------------------------------------------------
// Row softmax over s (length 1024), in place. One block per row.
// (Reference applies NO 1/sqrt(hd) scaling — kept faithful.)
// ------------------------------------------------------------------
__global__ void gqa_softmax(float* __restrict__ w) {
  float* p = w + (size_t)blockIdx.x * LL;
  __shared__ float red[256];
  const int tid = threadIdx.x;

  float m = -INFINITY;
  for (int i = tid; i < LL; i += 256) m = fmaxf(m, p[i]);
  red[tid] = m; __syncthreads();
  for (int st = 128; st > 0; st >>= 1) {
    if (tid < st) red[tid] = fmaxf(red[tid], red[tid + st]);
    __syncthreads();
  }
  m = red[0]; __syncthreads();

  float sum = 0.f;
  for (int i = tid; i < LL; i += 256) { float e = __expf(p[i] - m); p[i] = e; sum += e; }
  red[tid] = sum; __syncthreads();
  for (int st = 128; st > 0; st >>= 1) {
    if (tid < st) red[tid] += red[tid + st];
    __syncthreads();
  }
  const float inv = 1.0f / red[0];
  for (int i = tid; i < LL; i += 256) p[i] *= inv;
}

// ------------------------------------------------------------------
// ctx_bf[b,t,h*64+d] = sum_s weights[bh,t,s] * vt[(b*KVW+g*64+d)*LL+s]
// Per-wave: 16(t) x 64(d) = one whole head; causal-truncated contraction.
// ------------------------------------------------------------------
__global__ void gqa_ctx(const float* __restrict__ w, const __bf16* __restrict__ vt,
                        __bf16* __restrict__ ctxb) {
  const int lane  = threadIdx.x & 31;
  const int wave  = threadIdx.x >> 5;                 // 0..3
  const int tTile = blockIdx.y * 4 + wave;            // 0..63
  const int bh    = blockIdx.z;                       // 0..127
  const int b     = bh >> 5;
  const int h     = bh & 31;
  const int voff  = (h >> 2) * HD;                    // g*HD
  const int r16   = lane & 15;
  const int khalf = lane >> 4;
  const int t     = tTile * 16 + r16;

  v8f acc[4] = {};
  const int sMax = tTile * 16 + 16;                   // weights zero for s > t
  for (int kk = 0; kk < sMax; kk += 32) {
    Frag a, bfr[4];
    const float* ap = w + ((size_t)bh * LL + t) * LL + kk + khalf * 16;
    #pragma unroll
    for (int j = 0; j < 8; ++j) {
      a.h[j]     = (__bf16)ap[j];
      a.h[8 + j] = (__bf16)ap[8 + j];
    }
    const int k0 = kk + khalf * 16;
    #pragma unroll
    for (int j = 0; j < 4; ++j) {
      const __bf16* bp = vt + ((size_t)(b * KVW) + voff + j * 16 + r16) * LL + k0;
      bfr[j].q[0] = *(const u16x8*)bp;
      bfr[j].q[1] = *(const u16x8*)(bp + 8);
    }
    #pragma unroll
    for (int j = 0; j < 4; ++j)
      acc[j] = WMMA_BF16(a.v, bfr[j].v, acc[j]);
  }
  #pragma unroll
  for (int j = 0; j < 4; ++j) {
    const int cn = h * HD + j * 16 + r16;
    #pragma unroll
    for (int i = 0; i < 8; ++i) {
      const int tr = tTile * 16 + 8 * khalf + i;
      ctxb[((size_t)(b * LL + tr)) * DOUT + cn] = (__bf16)acc[j][i];
    }
  }
}

// ------------------------------------------------------------------
extern "C" void kernel_launch(void* const* d_in, const int* in_sizes, int n_in,
                              void* d_out, int out_size, void* d_ws, size_t ws_size,
                              hipStream_t stream) {
  (void)in_sizes; (void)n_in; (void)out_size; (void)ws_size;

  const float* Xq = (const float*)d_in[0];
  const float* Xk = (const float*)d_in[1];
  const float* Xv = (const float*)d_in[2];
  const float* Wq = (const float*)d_in[3];
  const float* Wk = (const float*)d_in[4];
  const float* Wv = (const float*)d_in[5];
  const float* Wo = (const float*)d_in[6];
  // d_in[7]: attn_mask — causal, computed analytically.

  float* attn_out = (float*)d_out;                          // B*L*DOUT fp32
  float* weights  = attn_out + (size_t)BB * LL * DOUT;      // B*H*L*L fp32

  // workspace (48 MB total)
  char* ws = (char*)d_ws;
  __bf16* xb  = (__bf16*)(ws);                              // 16 MB: Xq/Xk/Xv bf16
  __bf16* wt  = (__bf16*)(ws + (16u << 20));                //  8 MB: W^T bf16 (reused)
  __bf16* qcb = (__bf16*)(ws + (24u << 20));                // 16 MB: q_bf, later ctx_bf
  __bf16* kb  = (__bf16*)(ws + (40u << 20));                //  4 MB: k_bf
  __bf16* vt  = (__bf16*)(ws + (44u << 20));                //  4 MB: v^T bf16

  const int M = BB * LL;  // 4096
  const dim3 b256(256), b128(128);
  const dim3 gCvtX(M * DIN / (256 * 8));

  // ---- Q projection: q_bf[M, DOUT] ----
  gqa_cvt_bf16     <<<gCvtX, b256, 0, stream>>>(Xq, xb);
  gqa_transpose_cvt<<<dim3(DOUT / 32, DIN / 32), b256, 0, stream>>>(Wq, wt, DIN, DOUT);
  gqa_gemm         <<<dim3(DOUT / 256, M / 32), b128, 0, stream>>>(xb, wt, nullptr, qcb, M, DOUT, DIN, 1);

  // ---- K projection: k_bf[M, KVW] ----
  gqa_cvt_bf16     <<<gCvtX, b256, 0, stream>>>(Xk, xb);
  gqa_transpose_cvt<<<dim3(KVW / 32, DIN / 32), b256, 0, stream>>>(Wk, wt, DIN, KVW);
  gqa_gemm         <<<dim3(KVW / 256, M / 32), b128, 0, stream>>>(xb, wt, nullptr, kb, M, KVW, DIN, 1);

  // ---- V projection: vt[(b*KVW+n)*LL+s] ----
  gqa_cvt_bf16     <<<gCvtX, b256, 0, stream>>>(Xv, xb);
  gqa_transpose_cvt<<<dim3(KVW / 32, DIN / 32), b256, 0, stream>>>(Wv, wt, DIN, KVW);
  gqa_gemm         <<<dim3(KVW / 256, M / 32), b128, 0, stream>>>(xb, wt, nullptr, vt, M, KVW, DIN, 2);

  // ---- attention ----
  gqa_scores <<<dim3(2, LL / 16, BB * HH), b256, 0, stream>>>(qcb, kb, weights);
  gqa_softmax<<<dim3(BB * HH * LL),        b256, 0, stream>>>(weights);
  gqa_ctx    <<<dim3(1, 16, BB * HH),      b128, 0, stream>>>(weights, vt, qcb);  // q dead -> alias

  // ---- output projection: attn_out = ctx @ Wo (fp32 out) ----
  gqa_transpose_cvt<<<dim3(DOUT / 32, DOUT / 32), b256, 0, stream>>>(Wo, wt, DOUT, DOUT);
  gqa_gemm         <<<dim3(DOUT / 256, M / 32), b128, 0, stream>>>(qcb, wt, attn_out, nullptr, M, DOUT, DOUT, 0);
}
